// PointNetEncoder_25529285607669
// MI455X (gfx1250) — matrix operand order repaired and checked by reading, not compiled, not run on hardware
//
#include <hip/hip_runtime.h>
#include <hip/hip_fp16.h>

typedef __attribute__((ext_vector_type(16))) _Float16 v16h;
typedef __attribute__((ext_vector_type(8)))  _Float16 v8h;
typedef __attribute__((ext_vector_type(8)))  float    v8f;

#define NB      32      // point clouds
#define NPTS    2048    // points per cloud
#define KNN     16      // neighbors (incl. self)
#define TPB     8       // targets per block (one per wave32)
#define THREADS 256     // 8 waves
#define EPB     (TPB*KNN)  // 128 edges per block

// ---------------------------------------------------------------------------
// Kernel 1: KNN (register-resident top-16 per wave) + layer-1 edge MLP (WMMA)
//   writes: idxg[N*16] (u16 local neighbor idx), h1[N*32] (f16, post-ReLU)
// ---------------------------------------------------------------------------
__global__ __launch_bounds__(THREADS)
void knn_layer1(const float* __restrict__ pos,
                const float* __restrict__ w1a, const float* __restrict__ b1a,
                const float* __restrict__ w1b, const float* __restrict__ b1b,
                unsigned short* __restrict__ idxg, _Float16* __restrict__ h1)
{
    __shared__ float4   sp4[NPTS];       // 32 KB cloud (xyz0) -> ds_load_b128
    __shared__ int      knnS[TPB*KNN];
    __shared__ _Float16 Xe [EPB*32];     // edge feats (K padded 6->32)
    __shared__ _Float16 Yb [EPB*32];     // hidden after relu
    __shared__ _Float16 WaT[32*32];      // w1a transposed: WaT[n*32+k]
    __shared__ _Float16 WbT[32*32];      // w1b transposed

    const int tid  = threadIdx.x;
    const int lane = tid & 31;
    const int w    = tid >> 5;                 // wave id 0..7
    const int b    = blockIdx.x >> 8;          // 256 tiles per cloud
    const int tile = blockIdx.x & 255;

    // stage cloud + transposed weights
    for (int i = tid; i < NPTS; i += THREADS) {
        const float* p = pos + (size_t)(b*NPTS + i) * 3;
        sp4[i] = make_float4(p[0], p[1], p[2], 0.0f);
    }
    for (int i = tid; i < 32*32; i += THREADS) {
        int n = i >> 5, k = i & 31;
        WaT[i] = (_Float16)(k < 6 ? w1a[k*32 + n] : 0.0f);
        WbT[i] = (_Float16)w1b[k*32 + n];
    }
    __syncthreads();

    const int tl = tile*TPB + w;               // local target idx 0..2047
    const int gt = b*NPTS + tl;                // global point idx
    const float4 pi4 = sp4[tl];

    // --- all 2048 squared distances, 64 per lane in VGPRs ---
    float d2v[64];
    #pragma unroll
    for (int it = 0; it < 64; ++it) {
        float4 pj = sp4[it*32 + lane];
        float dx = pj.x - pi4.x, dy = pj.y - pi4.y, dz = pj.z - pi4.z;
        d2v[it] = dx*dx + dy*dy + dz*dz;
    }

    // --- 16 min-extraction passes; ties -> lowest index (top_k stability).
    //     Previous winner is invalidated inside the next scan (fused loop). ---
    int nb = 0, prev = -1;
    for (int p = 0; p < KNN; ++p) {
        float bv = 3.0e38f; int bj = NPTS;
        #pragma unroll
        for (int it = 0; it < 64; ++it) {
            int j = it*32 + lane;
            float v = d2v[it];
            if (j == prev) v = 3.0e38f;        // owner lane kills last winner
            d2v[it] = v;
            if (v < bv) { bv = v; bj = j; }    // within-lane: earliest j wins
        }
        #pragma unroll
        for (int off = 16; off > 0; off >>= 1) {
            float ov = __shfl_xor(bv, off, 32);
            int   oj = __shfl_xor(bj, off, 32);
            if (ov < bv || (ov == bv && oj < bj)) { bv = ov; bj = oj; }
        }
        if (lane == p) nb = bj;
        if (lane == 0) knnS[w*KNN + p] = bj;
        prev = bj;                              // wave-uniform
    }
    if (lane < KNN) idxg[(size_t)gt*KNN + lane] = (unsigned short)nb;
    __syncthreads();

    // --- edge features: [p_j (3), p_j - p_i (3), 0...] as f16, 2 threads/edge ---
    {
        int e  = tid >> 1;                 // 0..127
        int c0 = (tid & 1) * 16;
        int tt = e >> 4;
        int jl = knnS[tt*KNN + (e & 15)];
        int ti = tile*TPB + tt;
        float4 pj4 = sp4[jl];
        float4 pi_ = sp4[ti];
        float pj[3] = { pj4.x, pj4.y, pj4.z };
        float pi[3] = { pi_.x, pi_.y, pi_.z };
        v8h t0, t1;
        #pragma unroll
        for (int c = 0; c < 8; ++c) {
            int cc = c0 + c;
            float v = 0.0f;
            if (cc < 3)      v = pj[cc];
            else if (cc < 6) v = pj[cc-3] - pi[cc-3];
            t0[c] = (_Float16)v;
        }
        #pragma unroll
        for (int c = 0; c < 8; ++c) t1[c] = (_Float16)0.0f;  // cc >= 8 always 0
        *(v8h*)&Xe[e*32 + c0]     = t0;
        *(v8h*)&Xe[e*32 + c0 + 8] = t1;
    }
    __syncthreads();

    // --- GEMM 1a: [16x32] x [32x32] per wave, bias in acc, relu -> Yb ---
    const int m   = lane & 15;
    const int hb  = lane >> 4;             // half-wave selects K ranges (ISA 7.12.2)
    const int row = w*16 + m;

    v16h a;
    {
        v8h lo = *(const v8h*)&Xe[row*32 + hb*8];        // K 0-7 / 8-15
        v8h hi = *(const v8h*)&Xe[row*32 + 16 + hb*8];   // K 16-23 / 24-31
        #pragma unroll
        for (int i = 0; i < 8; ++i) { a[i] = lo[i]; a[8+i] = hi[i]; }
    }
    #pragma unroll
    for (int nt = 0; nt < 2; ++nt) {
        int n = (lane & 15) + nt*16;
        v16h bf;
        {
            v8h lo = *(const v8h*)&WaT[n*32 + hb*8];
            v8h hi = *(const v8h*)&WaT[n*32 + 16 + hb*8];
            #pragma unroll
            for (int i = 0; i < 8; ++i) { bf[i] = lo[i]; bf[8+i] = hi[i]; }
        }
        float bias = b1a[n];
        v8f c;
        #pragma unroll
        for (int v = 0; v < 8; ++v) c[v] = bias;
        c = __builtin_amdgcn_wmma_f32_16x16x32_f16(false, a, false, bf,
                                                   (short)0, c, false, false);
        #pragma unroll
        for (int v = 0; v < 8; ++v) {
            int mm = hb*8 + v;
            float r = c[v]; r = r > 0.f ? r : 0.f;
            Yb[(w*16 + mm)*32 + n] = (_Float16)r;
        }
    }
    __syncthreads();

    // --- GEMM 1b + max over this target's 16 edges (M dim) + outer ReLU ---
    v16h a2;
    {
        v8h lo = *(const v8h*)&Yb[row*32 + hb*8];
        v8h hi = *(const v8h*)&Yb[row*32 + 16 + hb*8];
        #pragma unroll
        for (int i = 0; i < 8; ++i) { a2[i] = lo[i]; a2[8+i] = hi[i]; }
    }
    #pragma unroll
    for (int nt = 0; nt < 2; ++nt) {
        int n = (lane & 15) + nt*16;
        v16h bf;
        {
            v8h lo = *(const v8h*)&WbT[n*32 + hb*8];
            v8h hi = *(const v8h*)&WbT[n*32 + 16 + hb*8];
            #pragma unroll
            for (int i = 0; i < 8; ++i) { bf[i] = lo[i]; bf[8+i] = hi[i]; }
        }
        float bias = b1b[n];
        v8f c;
        #pragma unroll
        for (int v = 0; v < 8; ++v) c[v] = bias;
        c = __builtin_amdgcn_wmma_f32_16x16x32_f16(false, a2, false, bf,
                                                   (short)0, c, false, false);
        float mx = c[0];
        #pragma unroll
        for (int v = 1; v < 8; ++v) mx = fmaxf(mx, c[v]);
        mx = fmaxf(mx, __shfl_xor(mx, 16, 32));   // combine row halves m / m+8
        mx = fmaxf(mx, 0.0f);                     // relu after layer 1
        if (lane < 16) h1[(size_t)gt*32 + n] = (_Float16)mx;
    }
}

// ---------------------------------------------------------------------------
// Kernel 2: layer-2 edge MLP (35->32->32, K padded to 64) + max pool -> out f32
// ---------------------------------------------------------------------------
__global__ __launch_bounds__(THREADS)
void layer2_kernel(const float* __restrict__ pos,
                   const float* __restrict__ w2a, const float* __restrict__ b2a,
                   const float* __restrict__ w2b, const float* __restrict__ b2b,
                   const unsigned short* __restrict__ idxg,
                   const _Float16* __restrict__ h1,
                   float* __restrict__ out)
{
    __shared__ _Float16 Xe [EPB*64];   // 16 KB edge feats (35 -> 64)
    __shared__ _Float16 Yb [EPB*32];
    __shared__ _Float16 WaT[32*64];    // w2a transposed: WaT[n*64+k], k padded 35->64
    __shared__ _Float16 WbT[32*32];

    const int tid  = threadIdx.x;
    const int lane = tid & 31;
    const int w    = tid >> 5;
    const int b    = blockIdx.x >> 8;
    const int tile = blockIdx.x & 255;

    for (int i = tid; i < 32*64; i += THREADS) {
        int n = i >> 6, k = i & 63;
        WaT[i] = (_Float16)(k < 35 ? w2a[k*32 + n] : 0.0f);
        if (i < 32*32) {
            int n2 = i >> 5, k2 = i & 31;
            WbT[i] = (_Float16)w2b[k2*32 + n2];
        }
    }

    // edge features: [h1_j (32), p_j - p_i (3), 0...] ; 2 threads per edge
    {
        int e  = tid >> 1;
        int c0 = (tid & 1) * 32;
        int tt = e >> 4;
        int tg = b*NPTS + tile*TPB + tt;
        int jl = idxg[(size_t)tg*KNN + (e & 15)];
        int jg = b*NPTS + jl;
        if (c0 == 0) {
            // 64B row copy: 4x b128
            const v8h* hr = (const v8h*)(h1 + (size_t)jg*32);
            v8h* xd = (v8h*)&Xe[e*64];
            #pragma unroll
            for (int q = 0; q < 4; ++q) xd[q] = hr[q];
        } else {
            v8h t0, tz;
            #pragma unroll
            for (int c = 0; c < 8; ++c) tz[c] = (_Float16)0.0f;
            t0 = tz;
            #pragma unroll
            for (int c = 0; c < 3; ++c)
                t0[c] = (_Float16)(pos[(size_t)jg*3 + c] - pos[(size_t)tg*3 + c]);
            v8h* xd = (v8h*)&Xe[e*64 + 32];
            xd[0] = t0; xd[1] = tz; xd[2] = tz; xd[3] = tz;
        }
    }
    __syncthreads();

    const int m   = lane & 15;
    const int hb  = lane >> 4;
    const int row = w*16 + m;
    const int tgw = b*NPTS + tile*TPB + w;

    // A fragments for both K-steps (K = 64)
    v16h aks[2];
    #pragma unroll
    for (int ks = 0; ks < 2; ++ks) {
        v8h lo = *(const v8h*)&Xe[row*64 + ks*32 + hb*8];
        v8h hi = *(const v8h*)&Xe[row*64 + ks*32 + 16 + hb*8];
        #pragma unroll
        for (int i = 0; i < 8; ++i) { aks[ks][i] = lo[i]; aks[ks][8+i] = hi[i]; }
    }

    // --- GEMM 2a: accumulate two K=32 steps, bias in acc, relu -> Yb ---
    #pragma unroll
    for (int nt = 0; nt < 2; ++nt) {
        int n = (lane & 15) + nt*16;
        float bias = b2a[n];
        v8f c;
        #pragma unroll
        for (int v = 0; v < 8; ++v) c[v] = bias;
        #pragma unroll
        for (int ks = 0; ks < 2; ++ks) {
            v16h bf;
            v8h lo = *(const v8h*)&WaT[n*64 + ks*32 + hb*8];
            v8h hi = *(const v8h*)&WaT[n*64 + ks*32 + 16 + hb*8];
            #pragma unroll
            for (int i = 0; i < 8; ++i) { bf[i] = lo[i]; bf[8+i] = hi[i]; }
            c = __builtin_amdgcn_wmma_f32_16x16x32_f16(false, aks[ks], false, bf,
                                                       (short)0, c, false, false);
        }
        #pragma unroll
        for (int v = 0; v < 8; ++v) {
            int mm = hb*8 + v;
            float r = c[v]; r = r > 0.f ? r : 0.f;
            Yb[(w*16 + mm)*32 + n] = (_Float16)r;
        }
    }
    __syncthreads();

    // --- GEMM 2b + max over K edges; no trailing relu ---
    v16h a2;
    {
        v8h lo = *(const v8h*)&Yb[row*32 + hb*8];
        v8h hi = *(const v8h*)&Yb[row*32 + 16 + hb*8];
        #pragma unroll
        for (int i = 0; i < 8; ++i) { a2[i] = lo[i]; a2[8+i] = hi[i]; }
    }
    #pragma unroll
    for (int nt = 0; nt < 2; ++nt) {
        int n = (lane & 15) + nt*16;
        v16h bf;
        {
            v8h lo = *(const v8h*)&WbT[n*32 + hb*8];
            v8h hi = *(const v8h*)&WbT[n*32 + 16 + hb*8];
            #pragma unroll
            for (int i = 0; i < 8; ++i) { bf[i] = lo[i]; bf[8+i] = hi[i]; }
        }
        float bias = b2b[n];
        v8f c;
        #pragma unroll
        for (int v = 0; v < 8; ++v) c[v] = bias;
        c = __builtin_amdgcn_wmma_f32_16x16x32_f16(false, a2, false, bf,
                                                   (short)0, c, false, false);
        float mx = c[0];
        #pragma unroll
        for (int v = 1; v < 8; ++v) mx = fmaxf(mx, c[v]);
        mx = fmaxf(mx, __shfl_xor(mx, 16, 32));
        if (lane < 16) out[(size_t)tgw*32 + n] = mx;
    }
}

// ---------------------------------------------------------------------------
extern "C" void kernel_launch(void* const* d_in, const int* in_sizes, int n_in,
                              void* d_out, int out_size, void* d_ws, size_t ws_size,
                              hipStream_t stream)
{
    const float* pos = (const float*)d_in[0];
    // d_in[1] = batch (int64) is implicit in the [B, NPTS] layout; unused.
    const float* w1a = (const float*)d_in[2];
    const float* b1a = (const float*)d_in[3];
    const float* w1b = (const float*)d_in[4];
    const float* b1b = (const float*)d_in[5];
    const float* w2a = (const float*)d_in[6];
    const float* b2a = (const float*)d_in[7];
    const float* w2b = (const float*)d_in[8];
    const float* b2b = (const float*)d_in[9];
    float* out = (float*)d_out;

    // workspace: [0, 2MB) knn idx u16 ; [2MB, 6MB) h1 f16
    unsigned short* idxg = (unsigned short*)d_ws;
    _Float16* h1 = (_Float16*)((char*)d_ws + (size_t)NB*NPTS*KNN*sizeof(unsigned short));

    const int blocks = NB * (NPTS / TPB);   // 8192
    knn_layer1<<<blocks, THREADS, 0, stream>>>(pos, w1a, b1a, w1b, b1b, idxg, h1);
    layer2_kernel<<<blocks, THREADS, 0, stream>>>(pos, w2a, b2a, w2b, b2b, idxg, h1, out);
}
